// GQAttention_54485955117114
// MI455X (gfx1250) — matrix operand order repaired and checked by reading, not compiled, not run on hardware
//
#include <hip/hip_runtime.h>
#include <hip/hip_bf16.h>

// ---------------------------------------------------------------------------
// GQA attention (B=2, L=2048, H=2048, NH=32, NKV=8, HD=64) for gfx1250.
// f16 WMMA (v_wmma_f32_16x16x32_f16) everywhere, fp32 accumulation.
// Ping-pong double-buffered K-loop in GEMM (no rotation copies); async
// global->LDS staging of the V block in attention when available.
// ---------------------------------------------------------------------------

typedef __attribute__((ext_vector_type(16))) _Float16 v16h;
typedef __attribute__((ext_vector_type(8)))  float    v8f;
typedef __attribute__((ext_vector_type(4)))  unsigned int u32x4;
typedef __attribute__((ext_vector_type(4)))  int         i32x4;

// Address-space-qualified vector pointees for the gfx1250 async builtin:
// param0 is 'int4 addrspace(1)*' (prints as '__device__' in HIP diagnostics),
// param1 is the LDS-side pointer (addrspace(3)).
typedef __attribute__((address_space(1))) i32x4 gi32x4;
typedef __attribute__((address_space(3))) i32x4 li32x4;

#define B_   2
#define L_   2048
#define H_   2048
#define NH_  32
#define NKV_ 8
#define HD_  64
#define MTOK (B_ * L_)   // 4096 rows

#if defined(__has_builtin)
#if __has_builtin(__builtin_amdgcn_global_load_async_to_lds_b128) && \
    __has_builtin(__builtin_amdgcn_s_wait_asynccnt)
#define USE_ASYNC_COPY 1
#endif
#endif

union HFrag { v16h h; u32x4 q[2]; };

__device__ inline v8f zero8() {
  v8f z;
#pragma unroll
  for (int i = 0; i < 8; ++i) z[i] = 0.0f;
  return z;
}

// A fragment (16x32 f16, MxK). Lane l: row = l&15, two contiguous 8-half runs:
//   k = k0 + ((l&16)?8:0) + {0..7}   and   k = k0 + 16 + ((l&16)?8:0) + {0..7}
__device__ inline v16h load_fragA(const _Float16* base, int stride, int row0,
                                  int k0, int lane) {
  const int m   = lane & 15;
  const int off = (lane & 16) ? 8 : 0;
  const _Float16* p = base + (size_t)(row0 + m) * stride + k0 + off;
  HFrag f;
  f.q[0] = *reinterpret_cast<const u32x4*>(p);
  f.q[1] = *reinterpret_cast<const u32x4*>(p + 16);
  return f.h;
}

// B fragment (32x16 f16, KxN), element (k,n) = base[(col0+n)*stride + k].
// Lane l: n = l&15, two contiguous 8-half runs:
//   k = k0 + ((l&16)?8:0) + {0..7}   and   k = k0 + 16 + ((l&16)?8:0) + {0..7}
__device__ inline v16h load_fragB(const _Float16* base, int stride, int col0,
                                  int k0, int lane) {
  const int n    = lane & 15;
  const int koff = (lane & 16) ? 8 : 0;
  const _Float16* p = base + (size_t)(col0 + n) * stride + k0 + koff;
  HFrag f;
  f.q[0] = *reinterpret_cast<const u32x4*>(p);
  f.q[1] = *reinterpret_cast<const u32x4*>(p + 16);
  return f.h;
}

// ---------------------------------------------------------------------------
// 1) fp32 -> fp16 conversion
// ---------------------------------------------------------------------------
__global__ void cvt_f32_f16(const float* __restrict__ in,
                            _Float16* __restrict__ out, int n) {
  int i = blockIdx.x * blockDim.x + threadIdx.x;
  if (i < n) out[i] = (_Float16)in[i];
}

// ---------------------------------------------------------------------------
// 2) GEMM: Y(MxN) = X(MxK) * W(NxK)^T, f16 in, f16 or f32 out.
//    Block: 256 threads = 8 waves, tile 128x64. Wave: 16x64 strip.
//    Ping-pong double buffering (requires K % 64 == 0, true for all uses):
//    buffer A and buffer B alternate as load-target / compute-source, so the
//    scheduler overlaps VMEM latency with WMMAs without rotation copies.
// ---------------------------------------------------------------------------
__global__ __launch_bounds__(256) void gemm_wmma(
    const _Float16* __restrict__ X, const _Float16* __restrict__ W,
    _Float16* __restrict__ Y16, float* __restrict__ Y32,
    int M, int N, int K, int out32) {
  const int lane = threadIdx.x & 31;
  const int wave = threadIdx.x >> 5;
  const int m0 = blockIdx.y * 128 + wave * 16;
  const int n0 = blockIdx.x * 64;

  v8f acc[4];
#pragma unroll
  for (int t = 0; t < 4; ++t) acc[t] = zero8();

  v16h aA = load_fragA(X, K, m0, 0, lane);
  v16h bA[4];
#pragma unroll
  for (int t = 0; t < 4; ++t) bA[t] = load_fragB(W, K, n0 + t * 16, 0, lane);

  const int NS = K >> 5;                 // number of 32-wide K steps (even)
  int kk = 32;
  for (int it = 0; it < (NS - 2) / 2; ++it, kk += 64) {
    // Prefetch the K+128 stream into L2 (speculative, no counter cost).
    __builtin_prefetch(X + (size_t)(m0 + (lane & 15)) * K + kk + 128, 0, 1);
    __builtin_prefetch(W + (size_t)(n0 + (lane & 15) + ((lane >> 4) << 4)) * K
                           + kk + 128, 0, 1);
    // Load step kk into buffer B, compute step kk-32 from buffer A.
    v16h aB = load_fragA(X, K, m0, kk, lane);
    v16h bB[4];
#pragma unroll
    for (int t = 0; t < 4; ++t)
      bB[t] = load_fragB(W, K, n0 + t * 16, kk, lane);
#pragma unroll
    for (int t = 0; t < 4; ++t)
      acc[t] = __builtin_amdgcn_wmma_f32_16x16x32_f16(
          false, aA, false, bA[t], (short)0, acc[t], false, false);
    // Load step kk+32 into buffer A, compute step kk from buffer B.
    aA = load_fragA(X, K, m0, kk + 32, lane);
#pragma unroll
    for (int t = 0; t < 4; ++t)
      bA[t] = load_fragB(W, K, n0 + t * 16, kk + 32, lane);
#pragma unroll
    for (int t = 0; t < 4; ++t)
      acc[t] = __builtin_amdgcn_wmma_f32_16x16x32_f16(
          false, aB, false, bB[t], (short)0, acc[t], false, false);
  }
  // Peeled final pair: steps K-64 (in buffer A) and K-32.
  {
    v16h aB = load_fragA(X, K, m0, K - 32, lane);
    v16h bB[4];
#pragma unroll
    for (int t = 0; t < 4; ++t)
      bB[t] = load_fragB(W, K, n0 + t * 16, K - 32, lane);
#pragma unroll
    for (int t = 0; t < 4; ++t)
      acc[t] = __builtin_amdgcn_wmma_f32_16x16x32_f16(
          false, aA, false, bA[t], (short)0, acc[t], false, false);
#pragma unroll
    for (int t = 0; t < 4; ++t)
      acc[t] = __builtin_amdgcn_wmma_f32_16x16x32_f16(
          false, aB, false, bB[t], (short)0, acc[t], false, false);
  }

  const int moff = 8 * (lane >> 4);
  const int nlan = lane & 15;
#pragma unroll
  for (int t = 0; t < 4; ++t) {
#pragma unroll
    for (int r = 0; r < 8; ++r) {
      size_t row = (size_t)(m0 + r + moff);
      size_t col = (size_t)(n0 + t * 16 + nlan);
      if (out32) Y32[row * N + col] = acc[t][r];
      else       Y16[row * N + col] = (_Float16)acc[t][r];
    }
  }
}

// ---------------------------------------------------------------------------
// 3) RoPE (in-place on f16 q/k), optional scale folded in (1/sqrt(HD) for q).
// ---------------------------------------------------------------------------
__global__ void rope_kernel(_Float16* __restrict__ buf, int nheads, int stride,
                            float scale, int total) {
  int idx = blockIdx.x * blockDim.x + threadIdx.x;
  if (idx >= total) return;
  int i    = idx & 31;           // frequency index 0..31
  int rest = idx >> 5;
  int hh   = rest % nheads;
  int tok  = rest / nheads;      // 0..B*L-1
  int l    = tok % L_;           // sequence position
  // inv_freq = 10000^(-i/32) = exp(-i * ln(10000)/32)
  float f   = __expf(-(float)i * (9.210340371976184f / 32.0f));
  float ang = (float)l * f;
  float s, c;
  __sincosf(ang, &s, &c);
  size_t base = (size_t)tok * stride + (size_t)hh * HD_;
  float x1 = (float)buf[base + i];
  float x2 = (float)buf[base + i + 32];
  buf[base + i]      = (_Float16)((x1 * c - x2 * s) * scale);
  buf[base + i + 32] = (_Float16)((x2 * c + x1 * s) * scale);
}

// ---------------------------------------------------------------------------
// 4) Flash attention, causal, GQA (4 q-heads share one kv-head).
//    Grid: (L/64, NH, B). Block: 128 threads = 4 waves; wave owns 16 q rows.
// ---------------------------------------------------------------------------
__global__ __launch_bounds__(128) void attn_kernel(
    const _Float16* __restrict__ qh,   // (B*L, NH*HD)
    const _Float16* __restrict__ kh,   // (B*L, NKV*HD)
    const _Float16* __restrict__ vh,   // (B*L, NKV*HD)
    _Float16* __restrict__ aoh) {      // (B*L, NH*HD)
  __shared__ __align__(16) _Float16 Vt[HD_ * 64];        // Vt[d][key]
  __shared__ __align__(16) _Float16 Pb[4][16 * 64];      // per-wave P tile
#ifdef USE_ASYNC_COPY
  __shared__ __align__(16) _Float16 Vraw[64 * HD_];      // Vraw[key][d]
#endif

  const int lane = threadIdx.x & 31;
  const int wave = threadIdx.x >> 5;
  const int qt = blockIdx.x;           // query tile (64 rows)
  const int h  = blockIdx.y;           // q head
  const int b  = blockIdx.z;
  const int g  = h / (NH_ / NKV_);     // kv head
  const int l0 = qt * 64;
  const int rowq0 = b * L_ + l0 + wave * 16;
  const int kvstride = NKV_ * HD_;     // 512

  // Q fragments (persist across key blocks); q already scaled by 1/sqrt(HD).
  v16h aq0 = load_fragA(qh, NH_ * HD_, rowq0, h * HD_ + 0,  lane);
  v16h aq1 = load_fragA(qh, NH_ * HD_, rowq0, h * HD_ + 32, lane);

  v8f o[4];
#pragma unroll
  for (int t = 0; t < 4; ++t) o[t] = zero8();
  float mrow[8], lrow[8];
#pragma unroll
  for (int r = 0; r < 8; ++r) { mrow[r] = -__builtin_inff(); lrow[r] = 0.0f; }

  const int moff = 8 * (lane >> 4);
  const int nlan = lane & 15;

  for (int kb = 0; kb <= qt; ++kb) {
    __syncthreads();

    // Prefetch next key block's K/V rows into L2 while we work on this one.
    if (kb < qt) {
      int pr = b * L_ + (kb + 1) * 64 + (threadIdx.x & 63);
      __builtin_prefetch(kh + (size_t)pr * kvstride + g * HD_, 0, 1);
      __builtin_prefetch(vh + (size_t)pr * kvstride + g * HD_, 0, 1);
    }

    // ---- Stage V block into LDS, transposed: Vt[d*64 + key] ----
#ifdef USE_ASYNC_COPY
    {
      const int tid = threadIdx.x;
#pragma unroll
      for (int c = 0; c < 4; ++c) {
        int chunk = tid + c * 128;          // 512 chunks of 8 halves (16 B)
        int key   = chunk >> 3;
        int d8    = (chunk & 7) * 8;
        const _Float16* src =
            vh + (size_t)(b * L_ + kb * 64 + key) * kvstride + g * HD_ + d8;
        _Float16* dst = &Vraw[key * HD_ + d8];
        __builtin_amdgcn_global_load_async_to_lds_b128(
            (gi32x4*)src, (li32x4*)dst, 0, 0);
      }
      __builtin_amdgcn_s_wait_asynccnt(0);
      __syncthreads();
      // Transpose LDS -> LDS (ds_load_b128 + ds_store_b16).
#pragma unroll
      for (int c = 0; c < 4; ++c) {
        int chunk = tid + c * 128;
        int key   = chunk >> 3;
        int d8    = (chunk & 7) * 8;
        HFrag f;
        f.q[0] = *reinterpret_cast<const u32x4*>(&Vraw[key * HD_ + d8]);
        f.q[1] = f.q[0];
#pragma unroll
        for (int j = 0; j < 8; ++j) Vt[(d8 + j) * 64 + key] = f.h[j];
      }
    }
#else
    {
      const int tid = threadIdx.x;
#pragma unroll
      for (int c = 0; c < 4; ++c) {
        int chunk = tid + c * 128;          // 512 chunks of 8 halves
        int key   = chunk >> 3;
        int d8    = (chunk & 7) * 8;
        const _Float16* src =
            vh + (size_t)(b * L_ + kb * 64 + key) * kvstride + g * HD_ + d8;
        u32x4 v = *reinterpret_cast<const u32x4*>(src);
        HFrag f; f.q[0] = v; f.q[1] = v;
#pragma unroll
        for (int j = 0; j < 8; ++j) Vt[(d8 + j) * 64 + key] = f.h[j];
      }
    }
#endif
    __syncthreads();

    // ---- S = Q * K^T (16x64 per wave): load all 8 B-fragments first, then
    //      issue the 8 WMMAs so VMEM latency overlaps the matrix pipe. ----
    v16h bk[4][2];
#pragma unroll
    for (int t = 0; t < 4; ++t) {
      bk[t][0] = load_fragB(kh, kvstride, b * L_ + kb * 64 + t * 16,
                            g * HD_ + 0, lane);
      bk[t][1] = load_fragB(kh, kvstride, b * L_ + kb * 64 + t * 16,
                            g * HD_ + 32, lane);
    }
    v8f s[4];
#pragma unroll
    for (int t = 0; t < 4; ++t) {
      s[t] = zero8();
      s[t] = __builtin_amdgcn_wmma_f32_16x16x32_f16(
          false, aq0, false, bk[t][0], (short)0, s[t], false, false);
      s[t] = __builtin_amdgcn_wmma_f32_16x16x32_f16(
          false, aq1, false, bk[t][1], (short)0, s[t], false, false);
    }

    // Causal mask (diagonal block only).
    if (kb == qt) {
#pragma unroll
      for (int t = 0; t < 4; ++t) {
        int key = kb * 64 + t * 16 + nlan;
#pragma unroll
        for (int r = 0; r < 8; ++r) {
          int qrow = l0 + wave * 16 + r + moff;
          if (key > qrow) s[t][r] = -__builtin_inff();
        }
      }
    }

    // Online softmax.
    float alpha[8];
#pragma unroll
    for (int r = 0; r < 8; ++r) {
      float mx = fmaxf(fmaxf(s[0][r], s[1][r]), fmaxf(s[2][r], s[3][r]));
      mx = fmaxf(mx, __shfl_xor(mx, 1, 32));
      mx = fmaxf(mx, __shfl_xor(mx, 2, 32));
      mx = fmaxf(mx, __shfl_xor(mx, 4, 32));
      mx = fmaxf(mx, __shfl_xor(mx, 8, 32));
      float mnew = fmaxf(mrow[r], mx);
      alpha[r] = __expf(mrow[r] - mnew);
      mrow[r] = mnew;
    }
#pragma unroll
    for (int t = 0; t < 4; ++t)
#pragma unroll
      for (int r = 0; r < 8; ++r)
        s[t][r] = __expf(s[t][r] - mrow[r]);
#pragma unroll
    for (int r = 0; r < 8; ++r) {
      float sm = s[0][r] + s[1][r] + s[2][r] + s[3][r];
      sm += __shfl_xor(sm, 1, 32);
      sm += __shfl_xor(sm, 2, 32);
      sm += __shfl_xor(sm, 4, 32);
      sm += __shfl_xor(sm, 8, 32);
      lrow[r] = lrow[r] * alpha[r] + sm;
    }
#pragma unroll
    for (int t = 0; t < 4; ++t)
#pragma unroll
      for (int r = 0; r < 8; ++r) o[t][r] *= alpha[r];

    // P (f16) via LDS to convert C-layout -> A-layout.
#pragma unroll
    for (int t = 0; t < 4; ++t)
#pragma unroll
      for (int r = 0; r < 8; ++r)
        Pb[wave][(r + moff) * 64 + t * 16 + nlan] = (_Float16)s[t][r];
    __syncthreads();

    v16h ap0 = load_fragA(&Pb[wave][0], 64, 0, 0,  lane);
    v16h ap1 = load_fragA(&Pb[wave][0], 64, 0, 32, lane);
    v16h bv[4][2];
#pragma unroll
    for (int t = 0; t < 4; ++t) {
      bv[t][0] = load_fragB(Vt, 64, t * 16, 0,  lane);
      bv[t][1] = load_fragB(Vt, 64, t * 16, 32, lane);
    }
#pragma unroll
    for (int t = 0; t < 4; ++t) {
      o[t] = __builtin_amdgcn_wmma_f32_16x16x32_f16(
          false, ap0, false, bv[t][0], (short)0, o[t], false, false);
      o[t] = __builtin_amdgcn_wmma_f32_16x16x32_f16(
          false, ap1, false, bv[t][1], (short)0, o[t], false, false);
    }
  }

  // Normalize and store (B*L, NH*HD) f16.
#pragma unroll
  for (int r = 0; r < 8; ++r) lrow[r] = 1.0f / lrow[r];
#pragma unroll
  for (int t = 0; t < 4; ++t)
#pragma unroll
    for (int r = 0; r < 8; ++r) {
      size_t row = (size_t)(rowq0 + r + moff);
      size_t col = (size_t)(h * HD_ + t * 16 + nlan);
      aoh[row * (NH_ * HD_) + col] = (_Float16)(o[t][r] * lrow[r]);
    }
}

// ---------------------------------------------------------------------------
// Host-side launch
// ---------------------------------------------------------------------------
extern "C" void kernel_launch(void* const* d_in, const int* in_sizes, int n_in,
                              void* d_out, int out_size, void* d_ws,
                              size_t ws_size, hipStream_t stream) {
  const float* x  = (const float*)d_in[0];
  const float* wq = (const float*)d_in[1];
  const float* wk = (const float*)d_in[2];
  const float* wv = (const float*)d_in[3];
  const float* wo = (const float*)d_in[4];
  float* out = (float*)d_out;

  const int nX  = MTOK * H_;          // 8,388,608
  const int nWq = NH_ * HD_ * H_;     // 4,194,304
  const int nWk = NKV_ * HD_ * H_;    // 1,048,576
  const int nWo = H_ * NH_ * HD_;     // 4,194,304
  const int nQ  = MTOK * NH_ * HD_;   // 8,388,608
  const int nK  = MTOK * NKV_ * HD_;  // 2,097,152

  _Float16* xh  = (_Float16*)d_ws;
  _Float16* wqh = xh  + nX;
  _Float16* wkh = wqh + nWq;
  _Float16* wvh = wkh + nWk;
  _Float16* woh = wvh + nWk;
  _Float16* qh  = woh + nWo;
  _Float16* kh  = qh  + nQ;
  _Float16* vh  = kh  + nK;
  _Float16* aoh = vh  + nK;

  // 1) convert to f16
  cvt_f32_f16<<<(nX  + 255) / 256, 256, 0, stream>>>(x,  xh,  nX);
  cvt_f32_f16<<<(nWq + 255) / 256, 256, 0, stream>>>(wq, wqh, nWq);
  cvt_f32_f16<<<(nWk + 255) / 256, 256, 0, stream>>>(wk, wkh, nWk);
  cvt_f32_f16<<<(nWk + 255) / 256, 256, 0, stream>>>(wv, wvh, nWk);
  cvt_f32_f16<<<(nWo + 255) / 256, 256, 0, stream>>>(wo, woh, nWo);

  // 2) QKV projections (f16 out)
  gemm_wmma<<<dim3((NH_ * HD_) / 64, MTOK / 128), 256, 0, stream>>>(
      xh, wqh, qh, nullptr, MTOK, NH_ * HD_, H_, 0);
  gemm_wmma<<<dim3((NKV_ * HD_) / 64, MTOK / 128), 256, 0, stream>>>(
      xh, wkh, kh, nullptr, MTOK, NKV_ * HD_, H_, 0);
  gemm_wmma<<<dim3((NKV_ * HD_) / 64, MTOK / 128), 256, 0, stream>>>(
      xh, wvh, vh, nullptr, MTOK, NKV_ * HD_, H_, 0);

  // 3) RoPE (scale 1/sqrt(HD) folded into q)
  {
    int totq = MTOK * NH_ * 32;
    int totk = MTOK * NKV_ * 32;
    rope_kernel<<<(totq + 255) / 256, 256, 0, stream>>>(
        qh, NH_, NH_ * HD_, 0.125f, totq);
    rope_kernel<<<(totk + 255) / 256, 256, 0, stream>>>(
        kh, NKV_, NKV_ * HD_, 1.0f, totk);
  }

  // 4) attention
  attn_kernel<<<dim3(L_ / 64, NH_, B_), 128, 0, stream>>>(qh, kh, vh, aoh);

  // 5) output projection (f32 out)
  gemm_wmma<<<dim3(H_ / 64, MTOK / 128), 256, 0, stream>>>(
      aoh, woh, nullptr, out, MTOK, H_, NH_ * HD_, 1);
}
